// StudentModel_90512140796081
// MI455X (gfx1250) — matrix operand order repaired and checked by reading, not compiled, not run on hardware
//
#include <hip/hip_runtime.h>
#include <hip/hip_bf16.h>
#include <stdint.h>

#define NN   50000
#define EE   800000
#define ETOT (EE + NN)
#define NEG_SLOPE 0.2f

typedef __attribute__((ext_vector_type(16))) __bf16 v16bf;
typedef __attribute__((ext_vector_type(8)))  float  v8f;

// ---------- helpers ----------
__device__ __forceinline__ unsigned short f2bf(float f) {
  unsigned u = __float_as_uint(f);
  unsigned r = u + 0x7FFFu + ((u >> 16) & 1u);   // round-to-nearest-even
  return (unsigned short)(r >> 16);
}
__device__ __forceinline__ unsigned ford(float f) {   // monotone float -> u32
  unsigned u = __float_as_uint(f);
  return (u & 0x80000000u) ? ~u : (u | 0x80000000u);
}
__device__ __forceinline__ float fuord(unsigned u) {  // inverse
  unsigned v = (u & 0x80000000u) ? (u & 0x7FFFFFFFu) : ~u;
  return __uint_as_float(v);
}

// ---------- fp32 -> bf16 convert ----------
__global__ void k_f32_to_bf16(const float* __restrict__ in,
                              unsigned short* __restrict__ out, int n) {
  int i = blockIdx.x * blockDim.x + threadIdx.x;
  if (i < n) out[i] = f2bf(in[i]);
}

// ---------- GEMM: C[M,Nc] = A[M,K](bf16) * B[K,Nc](bf16), f32 accumulate ----------
// Block = 256 threads = 8 waves. Block tile: 128 rows x 64 cols.
// Wave w owns rows [128*by + 16w, +16), computes 4 x 16x16 WMMA tiles (A reuse).
// A tile (128x32) and B tile (32x64, stored transposed+fragment-ordered) staged in LDS.
__global__ void __launch_bounds__(256)
k_gemm_wmma(const unsigned short* __restrict__ A,
            const unsigned short* __restrict__ B,
            float* __restrict__ C, int M, int K, int Nc) {
  __shared__ unsigned short As[128 * 32];   // [row][k]           8 KB
  __shared__ unsigned short Bt[64 * 32];    // [col][k] transposed 4 KB

  const int tid  = threadIdx.x;
  const int wave = tid >> 5;
  const int lane = tid & 31;
  const int half = lane >> 4;
  const int l16  = lane & 15;
  const int rowBase = blockIdx.y * 128;
  const int colBase = blockIdx.x * 64;

  v8f acc[4];
#pragma unroll
  for (int ct = 0; ct < 4; ++ct) acc[ct] = v8f{};

  for (int k0 = 0; k0 < K; k0 += 32) {
    // ---- stage A tile: thread t loads 16 bf16 (32B) of one half-row ----
    {
      int row = tid >> 1;               // 0..127
      int kh  = tid & 1;                // which 16-wide half of the 32 ks
      int grow = rowBase + row;
      uint4 v0 = {0, 0, 0, 0}, v1 = {0, 0, 0, 0};
      if (grow < M) {
        const uint4* s = (const uint4*)(A + (size_t)grow * K + k0 + kh * 16);
        v0 = s[0]; v1 = s[1];
      }
      uint4* d = (uint4*)(&As[row * 32 + kh * 16]);
      d[0] = v0; d[1] = v1;
    }
    // ---- stage B tile transposed: thread t loads 8 bf16 (16B) of one row ----
    {
      int brow = tid >> 3;              // k within tile: 0..31
      int bc   = (tid & 7) * 8;         // col within tile
      union { uint4 q; unsigned short us[8]; } bu;
      bu.q = *(const uint4*)(B + (size_t)(k0 + brow) * Nc + colBase + bc);
#pragma unroll
      for (int j = 0; j < 8; ++j) Bt[(bc + j) * 32 + brow] = bu.us[j];
    }
    __syncthreads();

    // ---- build A fragment (contiguous k = 16*half + s, matching B), reused for 4 tiles ----
    union { unsigned u32[8]; v16bf v; } a;
    const unsigned short* Ar = &As[(wave * 16 + l16) * 32];
#pragma unroll
    for (int g = 0; g < 8; ++g) {
      int k = (g << 1) + (half << 4);
      a.u32[g] = *(const unsigned*)(Ar + k);          // K pair (k, k+1)
    }
    // ---- 4 column tiles: B fragment = one contiguous 32B LDS read ----
#pragma unroll
    for (int ct = 0; ct < 4; ++ct) {
      v16bf bv = *(const v16bf*)(&Bt[(ct * 16 + l16) * 32 + half * 16]);
      acc[ct] = __builtin_amdgcn_wmma_f32_16x16x32_bf16(
          false, a.v, false, bv, (short)0, acc[ct], false, false);
    }
    __syncthreads();
  }

  // ---- store: C/D layout -> VGPR i holds (M = 16w + 8*half + i, N = col) ----
#pragma unroll
  for (int ct = 0; ct < 4; ++ct) {
    int col = colBase + ct * 16 + l16;
#pragma unroll
    for (int i = 0; i < 8; ++i) {
      int row = rowBase + wave * 16 + half * 8 + i;
      if (row < M) C[(size_t)row * Nc + col] = acc[ct][i];
    }
  }
}

// ---------- per-(node,head) attention scores ----------
__global__ void k_scores(const float* __restrict__ Hf, const float* __restrict__ a_src,
                         const float* __restrict__ a_dst, float* __restrict__ s_src,
                         float* __restrict__ s_dst, int H, int C) {
  int i = blockIdx.x * blockDim.x + threadIdx.x;   // n*H + h
  if (i >= NN * H) return;
  int n = i / H, h = i % H;
  const float* hp = Hf + ((size_t)n * H + h) * C;
  const float* as = a_src + h * C;
  const float* ad = a_dst + h * C;
  float ss = 0.f, sd = 0.f;
  for (int c = 0; c < C; ++c) { float v = hp[c]; ss += v * as[c]; sd += v * ad[c]; }
  s_src[i] = ss; s_dst[i] = sd;
}

// ---------- edge pass 1: leaky-relu score + segment max ----------
__global__ void k_edge_score_max(const long long* __restrict__ e_src,
                                 const long long* __restrict__ e_dst,
                                 const float* __restrict__ s_src,
                                 const float* __restrict__ s_dst,
                                 float* __restrict__ e_att,
                                 unsigned* __restrict__ m_ord, int H) {
  long long i = (long long)blockIdx.x * blockDim.x + threadIdx.x;
  long long tot = (long long)ETOT * H;
  if (i >= tot) return;
  int h = (int)(i % H);
  long long e = i / H;
  int s, d;
  if (e < EE) { s = (int)e_src[e]; d = (int)e_dst[e]; }
  else        { s = d = (int)(e - EE); }              // self loops
  float v = s_src[(size_t)s * H + h] + s_dst[(size_t)d * H + h];
  v = (v > 0.f) ? v : NEG_SLOPE * v;
  e_att[i] = v;
  atomicMax(&m_ord[(size_t)d * H + h], ford(v));
}

// ---------- edge pass 2: exp(e - m[dst]) + segment sum ----------
__global__ void k_edge_exp_sum(const long long* __restrict__ e_dst,
                               const float* __restrict__ e_att,
                               const unsigned* __restrict__ m_ord,
                               float* __restrict__ w_att,
                               float* __restrict__ denom, int H) {
  long long i = (long long)blockIdx.x * blockDim.x + threadIdx.x;
  long long tot = (long long)ETOT * H;
  if (i >= tot) return;
  int h = (int)(i % H);
  long long e = i / H;
  int d = (e < EE) ? (int)e_dst[e] : (int)(e - EE);
  float m = fuord(m_ord[(size_t)d * H + h]);
  float w = __expf(e_att[i] - m);
  w_att[i] = w;
  atomicAdd(&denom[(size_t)d * H + h], w);
}

// ---------- edge pass 3: weighted scatter-add (one wave per (edge,head)) ----------
__global__ void k_edge_aggregate(const long long* __restrict__ e_src,
                                 const long long* __restrict__ e_dst,
                                 const float* __restrict__ w_att,
                                 const float* __restrict__ denom,
                                 const float* __restrict__ Hf,
                                 float* __restrict__ acc, int H, int C) {
  long long wid = ((long long)blockIdx.x * blockDim.x + threadIdx.x) >> 5;
  int lane = threadIdx.x & 31;
  long long tot = (long long)ETOT * H;
  if (wid >= tot) return;
  int h = (int)(wid % H);
  long long e = wid / H;
  int s, d;
  if (e < EE) { s = (int)e_src[e]; d = (int)e_dst[e]; }
  else        { s = d = (int)(e - EE); }
  const float* hp = Hf  + ((size_t)s * H + h) * C;
  float*       op = acc + ((size_t)d * H + h) * C;
  __builtin_prefetch(hp + lane, 0, 0);                // global_prefetch_b8
  float alpha = w_att[wid] / (denom[(size_t)d * H + h] + 1e-16f);
  for (int c = lane; c < C; c += 32)                  // C==64 -> 2 coalesced iters
    atomicAdd(&op[c], alpha * hp[c]);
}

// ---------- finalize: ELU(acc + b) ----------
__global__ void k_finalize_elu(const float* __restrict__ acc,
                               const float* __restrict__ bias,
                               float* __restrict__ out, int F) {
  long long i = (long long)blockIdx.x * blockDim.x + threadIdx.x;
  if (i >= (long long)NN * F) return;
  float v = acc[i] + bias[(int)(i % F)];
  out[i] = (v > 0.f) ? v : (__expf(v) - 1.f);
}

// ---------- finalize layer 3: mean over heads + b3 ----------
__global__ void k_finalize_mean(const float* __restrict__ acc,
                                const float* __restrict__ bias,
                                float* __restrict__ out, int H, int C) {
  int i = blockIdx.x * blockDim.x + threadIdx.x;   // n*C + c
  if (i >= NN * C) return;
  int n = i / C, c = i % C;
  float s = 0.f;
  for (int h = 0; h < H; ++h) s += acc[((size_t)n * H + h) * C + c];
  out[i] = s / (float)H + bias[c];
}

// ---------- host orchestration ----------
extern "C" void kernel_launch(void* const* d_in, const int* in_sizes, int n_in,
                              void* d_out, int out_size, void* d_ws, size_t ws_size,
                              hipStream_t stream) {
  const float*     x    = (const float*)d_in[0];
  const long long* ei   = (const long long*)d_in[1];
  const long long* esrc = ei;
  const long long* edst = ei + EE;
  const float* W1  = (const float*)d_in[2];
  const float* a1s = (const float*)d_in[3];
  const float* a1d = (const float*)d_in[4];
  const float* b1  = (const float*)d_in[5];
  const float* W2  = (const float*)d_in[6];
  const float* a2s = (const float*)d_in[7];
  const float* a2d = (const float*)d_in[8];
  const float* b2  = (const float*)d_in[9];
  const float* W3  = (const float*)d_in[10];
  const float* a3s = (const float*)d_in[11];
  const float* a3d = (const float*)d_in[12];
  const float* b3  = (const float*)d_in[13];
  float* out = (float*)d_out;

  // carve workspace
  char* p = (char*)d_ws;
  auto carve = [&](size_t bytes) -> void* {
    void* r = (void*)p; p += (bytes + 255) & ~(size_t)255; return r;
  };
  unsigned short* Xbf  = (unsigned short*)carve((size_t)NN * 256 * 2);
  unsigned short* Wbf  = (unsigned short*)carve((size_t)256 * 384 * 2);
  float* Hf    = (float*)carve((size_t)NN * 384 * 4);
  float* accB  = (float*)carve((size_t)NN * 384 * 4);
  float* x1    = (float*)carve((size_t)NN * 256 * 4);
  float* x2    = (float*)carve((size_t)NN * 256 * 4);
  float* ssrc  = (float*)carve((size_t)NN * 6 * 4);
  float* sdst  = (float*)carve((size_t)NN * 6 * 4);
  unsigned* mord = (unsigned*)carve((size_t)NN * 6 * 4);
  float* denom = (float*)carve((size_t)NN * 6 * 4);
  float* eatt  = (float*)carve((size_t)ETOT * 6 * 4);
  float* watt  = (float*)carve((size_t)ETOT * 6 * 4);

  const int TPB = 256;
  auto blks = [](long long n, int t) { return (unsigned)((n + t - 1) / t); };

  auto run_layer = [&](const float* Xin, int Fin, const float* W,
                       const float* as_, const float* ad_, int H, int C) {
    int Fout = H * C;
    // 1) convert inputs + weights to bf16
    k_f32_to_bf16<<<blks((long long)NN * Fin, TPB), TPB, 0, stream>>>(Xin, Xbf, NN * Fin);
    k_f32_to_bf16<<<blks((long long)Fin * Fout, TPB), TPB, 0, stream>>>(W, Wbf, Fin * Fout);
    // 2) H = X @ W via LDS-staged WMMA (block tile 128x64)
    dim3 grid(Fout / 64, (NN + 127) / 128);
    k_gemm_wmma<<<grid, TPB, 0, stream>>>(Xbf, Wbf, Hf, NN, Fin, Fout);
    // 3) attention logits per node/head
    k_scores<<<blks((long long)NN * H, TPB), TPB, 0, stream>>>(Hf, as_, ad_, ssrc, sdst, H, C);
    // 4) segment softmax over edges
    hipMemsetAsync(mord, 0, (size_t)NN * H * 4, stream);    // ordered(-max) identity
    hipMemsetAsync(denom, 0, (size_t)NN * H * 4, stream);
    hipMemsetAsync(accB, 0, (size_t)NN * Fout * 4, stream);
    long long th = (long long)ETOT * H;
    k_edge_score_max<<<blks(th, TPB), TPB, 0, stream>>>(esrc, edst, ssrc, sdst, eatt, mord, H);
    k_edge_exp_sum<<<blks(th, TPB), TPB, 0, stream>>>(edst, eatt, mord, watt, denom, H);
    k_edge_aggregate<<<blks(th * 32, TPB), TPB, 0, stream>>>(esrc, edst, watt, denom, Hf, accB, H, C);
  };

  // Layer 1: 128 -> 4x64, concat, ELU
  run_layer(x, 128, W1, a1s, a1d, 4, 64);
  k_finalize_elu<<<blks((long long)NN * 256, TPB), TPB, 0, stream>>>(accB, b1, x1, 256);
  // Layer 2: 256 -> 4x64, concat, ELU
  run_layer(x1, 256, W2, a2s, a2d, 4, 64);
  k_finalize_elu<<<blks((long long)NN * 256, TPB), TPB, 0, stream>>>(accB, b2, x2, 256);
  // Layer 3: 256 -> 6x64, mean over heads + b3
  run_layer(x2, 256, W3, a3s, a3d, 6, 64);
  k_finalize_mean<<<blks((long long)NN * 64, TPB), TPB, 0, stream>>>(accB, b3, out, 6, 64);
}
